// Plamo2MambaMixer_19464791785814
// MI455X (gfx1250) — matrix operand order repaired and checked by reading, not compiled, not run on hardware
//
#include <hip/hip_runtime.h>
#include <hip/hip_bf16.h>

// ---------------------------------------------------------------------------
// Plamo2 Mamba mixer for MI455X (gfx1250, wave32, WMMA bf16, async->LDS)
// HSZ=4096, L=2048, I=8192, N=64, K=4, TSR=256, H=64, P=128
// ---------------------------------------------------------------------------

#define HSZ   4096
#define SEQL  2048
#define ISZ   8192
#define KCONV 4
#define NST   64
#define TSR   256
#define NH    64
#define PH    128
#define PROJW (2 * ISZ)          // 16384
#define SSMW  (TSR + 2 * NST)    // 384

typedef __attribute__((ext_vector_type(16))) __bf16 v16bf;
typedef __attribute__((ext_vector_type(8)))  float  v8f;

// ---------------------------------------------------------------------------
// fp32 -> bf16 cast (vectorized)
// ---------------------------------------------------------------------------
__global__ __launch_bounds__(256) void cast_f32_bf16(const float* __restrict__ src,
                                                     __bf16* __restrict__ dst, int n) {
    int idx = (blockIdx.x * 256 + threadIdx.x) * 4;
    if (idx < n) {
        float4 v = *reinterpret_cast<const float4*>(src + idx);
        dst[idx + 0] = (__bf16)v.x;
        dst[idx + 1] = (__bf16)v.y;
        dst[idx + 2] = (__bf16)v.z;
        dst[idx + 3] = (__bf16)v.w;
    }
}

// ---------------------------------------------------------------------------
// WMMA bf16 TN GEMM:  C[M,N] = A[M,K] * B[N,K]^T
// Block tile 256x128, BK=64, 8 waves (4Mx2N); each wave owns a 64x64 patch
// (4x4 WMMA tiles, 128 acc VGPRs) -> 32 WMMA per wave per stage against
// 16 fragment loads: 2.0 WMMA per ds_load_b128 pair.
// Global->LDS via GLOBAL_LOAD_ASYNC_TO_LDS_B128 (ASYNCcnt), double buffered.
// ---------------------------------------------------------------------------
union FragU {
    v16bf v;
    uint4 u[2];
};

__device__ inline v16bf load_frag_lds(const __bf16* p) {
    FragU f;
    f.u[0] = *reinterpret_cast<const uint4*>(p);
    f.u[1] = *reinterpret_cast<const uint4*>(p + 16);
    return f.v;
}

// lane-private LDS byte address: flat->LDS aperture mapping keeps the LDS
// offset in addr[31:0] (ISA 10.2), so truncation is the DS address.
__device__ inline uint32_t lds_addr32(const void* p) {
    return (uint32_t)(uintptr_t)p;
}

// 64 contiguous bytes: 4 async b128 loads (INST_OFFSET applies to both the
// global and LDS addresses per the ISA pseudocode)
__device__ inline void async_copy64(uint32_t lds, const __bf16* gptr) {
    uint64_t ga = (uint64_t)(uintptr_t)gptr;
    asm volatile(
        "global_load_async_to_lds_b128 %0, %1, off\n\t"
        "global_load_async_to_lds_b128 %0, %1, off offset:16\n\t"
        "global_load_async_to_lds_b128 %0, %1, off offset:32\n\t"
        "global_load_async_to_lds_b128 %0, %1, off offset:48"
        :: "v"(lds), "v"(ga) : "memory");
}

// 128 contiguous bytes: 8 async b128 loads
__device__ inline void async_copy128(uint32_t lds, const __bf16* gptr) {
    uint64_t ga = (uint64_t)(uintptr_t)gptr;
    asm volatile(
        "global_load_async_to_lds_b128 %0, %1, off\n\t"
        "global_load_async_to_lds_b128 %0, %1, off offset:16\n\t"
        "global_load_async_to_lds_b128 %0, %1, off offset:32\n\t"
        "global_load_async_to_lds_b128 %0, %1, off offset:48\n\t"
        "global_load_async_to_lds_b128 %0, %1, off offset:64\n\t"
        "global_load_async_to_lds_b128 %0, %1, off offset:80\n\t"
        "global_load_async_to_lds_b128 %0, %1, off offset:96\n\t"
        "global_load_async_to_lds_b128 %0, %1, off offset:112"
        :: "v"(lds), "v"(ga) : "memory");
}

template <typename OUT>
__global__ __launch_bounds__(256)
void gemm_bf16_tn(const __bf16* __restrict__ A, const __bf16* __restrict__ B,
                  OUT* __restrict__ C, int Kd, int lda, int ldb, int ldc) {
    constexpr int BM = 256, BN = 128, BK = 64;
    __shared__ __align__(16) __bf16 As[2][BM * BK];   // 2 x 32 KB
    __shared__ __align__(16) __bf16 Bs[2][BN * BK];   // 2 x 16 KB

    const int tid  = threadIdx.x;
    const int wid  = tid >> 5;
    const int lane = tid & 31;
    const int half = lane >> 4;       // K-half selector per WMMA fragment layout
    const int ml   = lane & 15;       // m (or n) within 16x16 tile
    const int wm   = (wid & 3) * 64;  // 4 wave rows
    const int wn   = (wid >> 2) * 64; // 2 wave cols
    const long bm  = (long)blockIdx.y * BM;
    const long bn  = (long)blockIdx.x * BN;

    v8f acc[4][4];
#pragma unroll
    for (int a = 0; a < 4; ++a)
#pragma unroll
        for (int b = 0; b < 4; ++b)
            acc[a][b] = v8f{0.f, 0.f, 0.f, 0.f, 0.f, 0.f, 0.f, 0.f};

    // global -> LDS mapping:
    //   A: thread owns one full 64-elem (128B) row of the 256-row tile
    //   B: thread owns half a row (32 elems, 64B) of the 128-row tile
    const int brow = tid >> 1;
    const int bseg = (tid & 1) * 32;
    const __bf16* gA = A + (size_t)(bm + tid) * lda;
    const __bf16* gB = B + (size_t)(bn + brow) * ldb + bseg;
    uint32_t sAaddr[2], sBaddr[2];
    sAaddr[0] = lds_addr32(&As[0][tid * BK]);
    sAaddr[1] = lds_addr32(&As[1][tid * BK]);
    sBaddr[0] = lds_addr32(&Bs[0][brow * BK + bseg]);
    sBaddr[1] = lds_addr32(&Bs[1][brow * BK + bseg]);

    auto issue = [&](int buf, int k0) {
        async_copy128(sAaddr[buf], gA + k0);  // 8 async instr / wave
        async_copy64(sBaddr[buf], gB + k0);   // 4 async instr / wave
    };

    issue(0, 0);
    int cur = 0;
    for (int k0 = 0; k0 < Kd; k0 += BK) {
        if (k0 + BK < Kd) {
            issue(cur ^ 1, k0 + BK);
            // 24 in flight; in-order completion => <=12 means current buffer
            // is fully resident while the prefetch stays outstanding
            asm volatile("s_wait_asynccnt 0xc" ::: "memory");
        } else {
            asm volatile("s_wait_asynccnt 0x0" ::: "memory");
        }
        __syncthreads();

        const __bf16* as = &As[cur][0];
        const __bf16* bs = &Bs[cur][0];
#pragma unroll
        for (int kk2 = 0; kk2 < 2; ++kk2) {
            const int kk = kk2 * 32;
            v16bf bfr[4], afr[4];
#pragma unroll
            for (int b = 0; b < 4; ++b)
                bfr[b] = load_frag_lds(&bs[(wn + b * 16 + ml) * BK + kk + half * 8]);
#pragma unroll
            for (int a = 0; a < 4; ++a)
                afr[a] = load_frag_lds(&as[(wm + a * 16 + ml) * BK + kk + half * 8]);
#pragma unroll
            for (int a = 0; a < 4; ++a)
#pragma unroll
                for (int b = 0; b < 4; ++b)
                    acc[a][b] = __builtin_amdgcn_wmma_f32_16x16x32_bf16(
                        false, afr[a], false, bfr[b], (short)0,
                        acc[a][b], false, false);
        }
        __syncthreads();
        cur ^= 1;
    }

    // epilogue: VGPR j holds M = j + half*8, N = ml
#pragma unroll
    for (int a = 0; a < 4; ++a) {
        const long row0 = bm + wm + a * 16 + half * 8;
#pragma unroll
        for (int b = 0; b < 4; ++b) {
            const long col = bn + wn + b * 16 + ml;
#pragma unroll
            for (int j = 0; j < 8; ++j)
                C[(size_t)(row0 + j) * ldc + col] = (OUT)acc[a][b][j];
        }
    }
}

// ---------------------------------------------------------------------------
// Causal depthwise conv (K=4) + SiLU.  x_in = proj[:, I:].T ; out xs[L,I] bf16
// ---------------------------------------------------------------------------
__global__ __launch_bounds__(256) void conv_silu_kernel(const __bf16* __restrict__ proj,
                                                        const float* __restrict__ conv_w,
                                                        __bf16* __restrict__ xs) {
    int idx = blockIdx.x * 256 + threadIdx.x;   // < L*I
    int t = idx >> 13;                          // / 8192
    int i = idx & (ISZ - 1);
    float acc = 0.f;
#pragma unroll
    for (int k = 0; k < KCONV; ++k) {
        int tt = t - (KCONV - 1) + k;
        if (tt >= 0)
            acc += conv_w[i * KCONV + k] *
                   (float)proj[(size_t)tt * PROJW + ISZ + i];
    }
    float s = acc / (1.f + __expf(-acc));
    xs[idx] = (__bf16)s;
}

// ---------------------------------------------------------------------------
// Fused rmsnorm of B (64), C (64), ts (256) segments of ssm_p row
// ---------------------------------------------------------------------------
__global__ __launch_bounds__(384) void rmsnorm_kernel(const float* __restrict__ ssm,
                                                      const float* __restrict__ bw,
                                                      const float* __restrict__ cw,
                                                      const float* __restrict__ dtw,
                                                      float* __restrict__ Bn,
                                                      float* __restrict__ Cn,
                                                      float* __restrict__ ts) {
    __shared__ float row[SSMW];
    __shared__ float rs[3];
    const int t = blockIdx.x, c = threadIdx.x;
    row[c] = ssm[(size_t)t * SSMW + c];
    __syncthreads();
    if (c == 0) {
        float s = 0.f;
        for (int j = 0; j < 64; ++j) s += row[j] * row[j];
        rs[0] = rsqrtf(s * (1.f / 64.f) + 1e-6f);
    } else if (c == 1) {
        float s = 0.f;
        for (int j = 64; j < 128; ++j) s += row[j] * row[j];
        rs[1] = rsqrtf(s * (1.f / 64.f) + 1e-6f);
    } else if (c == 2) {
        float s = 0.f;
        for (int j = 128; j < 384; ++j) s += row[j] * row[j];
        rs[2] = rsqrtf(s * (1.f / 256.f) + 1e-6f);
    }
    __syncthreads();
    float v = row[c];
    if (c < 64)       Bn[(size_t)t * 64 + c]          = v * rs[0] * bw[c];
    else if (c < 128) Cn[(size_t)t * 64 + (c - 64)]   = v * rs[1] * cw[c - 64];
    else              ts[(size_t)t * 256 + (c - 128)] = v * rs[2] * dtw[c - 128];
}

// ---------------------------------------------------------------------------
// delta = softplus(ts @ W_dt^T + dt_bias)   (L x H, tiny)
// ---------------------------------------------------------------------------
__global__ __launch_bounds__(256) void delta_kernel(const float* __restrict__ ts,
                                                    const float* __restrict__ W_dt,
                                                    const float* __restrict__ dt_bias,
                                                    float* __restrict__ delta) {
    int idx = blockIdx.x * 256 + threadIdx.x;   // < L*H
    int t = idx >> 6;
    int h = idx & (NH - 1);
    const float* tr = ts + (size_t)t * TSR;
    const float* wr = W_dt + (size_t)h * TSR;
    float acc = dt_bias[h];
    for (int r = 0; r < TSR; ++r) acc += tr[r] * wr[r];
    delta[idx] = (acc > 20.f) ? acc : __logf(1.f + __expf(acc));
}

// ---------------------------------------------------------------------------
// Selective scan.  One lane = one channel i; h[64], A-row[64] in VGPRs.
// B_t / C_t / delta_t broadcast through LDS per step.
// Fused epilogue: y = (ys + x*D) * silu(gate) -> bf16 (GEMM3 input)
// ---------------------------------------------------------------------------
__global__ __launch_bounds__(256) void scan_kernel(const float* __restrict__ Aptr,
                                                   const float* __restrict__ Dptr,
                                                   const float* __restrict__ Bn,
                                                   const float* __restrict__ Cn,
                                                   const float* __restrict__ delta,
                                                   const __bf16* __restrict__ xs,
                                                   const __bf16* __restrict__ proj,
                                                   __bf16* __restrict__ yout) {
    __shared__ float Bs[NST];
    __shared__ float Cs[NST];
    __shared__ float dh[2];
    const int tid = threadIdx.x;
    const int i = blockIdx.x * 256 + tid;       // channel
    const int hsel = tid >> 7;                  // which of the 2 heads this block spans

    float Ar[NST], h[NST];
#pragma unroll
    for (int n = 0; n < NST; ++n) {
        Ar[n] = Aptr[(size_t)i * NST + n];
        h[n] = 0.f;
    }
    const float Di = Dptr[i];

    for (int t = 0; t < SEQL; ++t) {
        if (tid < 64)        Bs[tid]       = Bn[(size_t)t * 64 + tid];
        else if (tid < 128)  Cs[tid - 64]  = Cn[(size_t)t * 64 + (tid - 64)];
        else if (tid < 130)  dh[tid - 128] = delta[(size_t)t * NH + blockIdx.x * 2 + (tid - 128)];
        __syncthreads();

        const float d  = dh[hsel];
        const float x  = (float)xs[(size_t)t * ISZ + i];
        const float dx = d * x;
        float y = 0.f;
#pragma unroll
        for (int n = 0; n < NST; ++n) {
            float dA = __expf(d * Ar[n]);
            h[n] = h[n] * dA + dx * Bs[n];
            y += h[n] * Cs[n];
        }
        const float g  = (float)proj[(size_t)t * PROJW + i];
        const float sg = g / (1.f + __expf(-g));
        yout[(size_t)t * ISZ + i] = (__bf16)((y + x * Di) * sg);
        __syncthreads();
    }
}

// ---------------------------------------------------------------------------
// Host side
// ---------------------------------------------------------------------------
extern "C" void kernel_launch(void* const* d_in, const int* in_sizes, int n_in,
                              void* d_out, int out_size, void* d_ws, size_t ws_size,
                              hipStream_t stream) {
    const float* hs      = (const float*)d_in[0];   // (L, HSZ)
    const float* W_in    = (const float*)d_in[1];   // (2I, HSZ)
    const float* conv_w  = (const float*)d_in[2];   // (I, K)
    const float* W_x     = (const float*)d_in[3];   // (384, I)
    const float* dt_ln_w = (const float*)d_in[4];   // (256,)
    const float* b_ln_w  = (const float*)d_in[5];   // (64,)
    const float* c_ln_w  = (const float*)d_in[6];   // (64,)
    const float* W_dt    = (const float*)d_in[7];   // (H, 256)
    const float* dt_bias = (const float*)d_in[8];   // (H,)
    const float* A       = (const float*)d_in[9];   // (I, N)
    const float* D       = (const float*)d_in[10];  // (I,)
    const float* W_out   = (const float*)d_in[11];  // (HSZ, I)
    float* out           = (float*)d_out;           // (L, HSZ)

    // workspace carve-out (256B aligned)
    char* w = (char*)d_ws;
    auto carve = [&](size_t bytes) {
        char* p = w;
        w += (bytes + 255) & ~(size_t)255;
        return p;
    };
    __bf16* Win_b  = (__bf16*)carve((size_t)PROJW * HSZ * 2);   // 128 MB
    __bf16* Wout_b = (__bf16*)carve((size_t)HSZ * ISZ * 2);     //  64 MB
    __bf16* Wx_b   = (__bf16*)carve((size_t)SSMW * ISZ * 2);    //   6 MB
    __bf16* hs_b   = (__bf16*)carve((size_t)SEQL * HSZ * 2);    //  16 MB
    __bf16* proj_b = (__bf16*)carve((size_t)SEQL * PROJW * 2);  //  64 MB
    __bf16* xs_b   = (__bf16*)carve((size_t)SEQL * ISZ * 2);    //  32 MB
    float*  ssm    = (float*) carve((size_t)SEQL * SSMW * 4);   //   3 MB
    float*  Bn     = (float*) carve((size_t)SEQL * NST * 4);
    float*  Cn     = (float*) carve((size_t)SEQL * NST * 4);
    float*  ts     = (float*) carve((size_t)SEQL * TSR * 4);
    float*  delta  = (float*) carve((size_t)SEQL * NH * 4);
    __bf16* y_b    = (__bf16*)carve((size_t)SEQL * ISZ * 2);    //  32 MB

    // 1) casts to bf16
    {
        int n;
        n = PROJW * HSZ; cast_f32_bf16<<<n / 4 / 256, 256, 0, stream>>>(W_in, Win_b, n);
        n = HSZ * ISZ;   cast_f32_bf16<<<n / 4 / 256, 256, 0, stream>>>(W_out, Wout_b, n);
        n = SSMW * ISZ;  cast_f32_bf16<<<n / 4 / 256, 256, 0, stream>>>(W_x, Wx_b, n);
        n = SEQL * HSZ;  cast_f32_bf16<<<n / 4 / 256, 256, 0, stream>>>(hs, hs_b, n);
    }

    // 2) proj = hs @ W_in^T   (2048 x 16384, K=4096) -> bf16
    {
        dim3 grid(PROJW / 128, SEQL / 256);
        gemm_bf16_tn<__bf16><<<grid, 256, 0, stream>>>(hs_b, Win_b, proj_b,
                                                       HSZ, HSZ, HSZ, PROJW);
    }

    // 3) conv + silu -> xs (L x I, bf16)
    conv_silu_kernel<<<(SEQL * ISZ) / 256, 256, 0, stream>>>(proj_b, conv_w, xs_b);

    // 4) ssm_p = xs @ W_x^T   (2048 x 384, K=8192) -> f32
    {
        dim3 grid(SSMW / 128, SEQL / 256);
        gemm_bf16_tn<float><<<grid, 256, 0, stream>>>(xs_b, Wx_b, ssm,
                                                      ISZ, ISZ, ISZ, SSMW);
    }

    // 5) rmsnorm B/C/ts
    rmsnorm_kernel<<<SEQL, 384, 0, stream>>>(ssm, b_ln_w, c_ln_w, dt_ln_w, Bn, Cn, ts);

    // 6) delta = softplus(ts @ W_dt^T + bias)
    delta_kernel<<<(SEQL * NH) / 256, 256, 0, stream>>>(ts, W_dt, dt_bias, delta);

    // 7) selective scan + gating fused -> y_b (bf16)
    scan_kernel<<<ISZ / 256, 256, 0, stream>>>(A, D, Bn, Cn, delta, xs_b, proj_b, y_b);

    // 8) out = y @ W_out^T    (2048 x 4096, K=8192) -> f32
    {
        dim3 grid(HSZ / 128, SEQL / 256);
        gemm_bf16_tn<float><<<grid, 256, 0, stream>>>(y_b, Wout_b, out,
                                                      ISZ, ISZ, ISZ, HSZ);
    }
}
